// ConvAttention_15968688406882
// MI455X (gfx1250) — matrix operand order repaired and checked by reading, not compiled, not run on hardware
//
#include <hip/hip_runtime.h>
#include <hip/hip_bf16.h>

// ---------------------------------------------------------------------------
// ConvAttention for MI455X (gfx1250, wave32, WMMA).
// pack weights (f32->f16, K-major) -> 3 software-pipelined implicit-GEMM convs
// (v_wmma_f32_16x16x32_f16, f32 accum) -> split-K QK^T logits -> fused
// partial-sum + softmax -> attn*V GEMM. Layouts chosen so attention GEMM
// fragments load directly from global memory as aligned b128s.
// ---------------------------------------------------------------------------

typedef _Float16 h16;
typedef __attribute__((ext_vector_type(16))) _Float16 v16h;
typedef __attribute__((ext_vector_type(8)))  _Float16 v8h;
typedef __attribute__((ext_vector_type(4)))  _Float16 v4h;
typedef __attribute__((ext_vector_type(8)))  float    v8f;
typedef __attribute__((ext_vector_type(4)))  float    v4f;

#define NB    8
#define CIN   64
#define COUT  128
#define LIN   18
#define HWDIM 1024

// --- WMMA fragment helpers (contiguous-K 16-bit layouts, wave32) --------------------
// A (16x32): lanes 0-15: K 0-15; lanes 16-31: K 16-31.
static __device__ __forceinline__ v16h frag_a16(const h16* p) {
  v8h lo = *(const v8h*)(p);
  v8h hi = *(const v8h*)(p + 8);
  v16h r;
#pragma unroll
  for (int e = 0; e < 8; ++e) { r[e] = lo[e]; r[e + 8] = hi[e]; }
  return r;
}
// B (32x16): lane holds one column N=lane&15; lanes 0-15: K 0-15, lanes 16-31: K 16-31.
static __device__ __forceinline__ v16h frag_b16(const h16* p) {
  v8h lo = *(const v8h*)(p);
  v8h hi = *(const v8h*)(p + 8);
  v16h r;
#pragma unroll
  for (int e = 0; e < 8; ++e) { r[e] = lo[e]; r[e + 8] = hi[e]; }
  return r;
}
static __device__ __forceinline__ v8f wmma16(v16h a, v16h b, v8f c) {
  return __builtin_amdgcn_wmma_f32_16x16x32_f16(false, a, false, b,
                                                (short)0, c, false, false);
}
static __device__ __forceinline__ void zero8(v8f& c) {
#pragma unroll
  for (int e = 0; e < 8; ++e) c[e] = 0.0f;
}

// --- Kernel 1: repack conv weights f32 -> f16, K ordered (tap, cin) --------
// wq/wk: (128,64,1,3,3) -> [cout][r*64+ci], K=576.  wv: (128,64,3,3,3), K=1728.
__global__ __launch_bounds__(256) void pack_weights(
    const float* __restrict__ wq, const float* __restrict__ wk,
    const float* __restrict__ wv, h16* __restrict__ wqp,
    h16* __restrict__ wkp, h16* __restrict__ wvp) {
  int id = blockIdx.x * 256 + threadIdx.x;
  if (id < 73728) {
    int co = id / 576, rem = id % 576, r = rem >> 6, ci = rem & 63;
    wqp[id] = (h16)wq[((size_t)co * 64 + ci) * 9 + r];
  } else if (id < 147456) {
    int id2 = id - 73728;
    int co = id2 / 576, rem = id2 % 576, r = rem >> 6, ci = rem & 63;
    wkp[id2] = (h16)wk[((size_t)co * 64 + ci) * 9 + r];
  } else if (id < 368640) {
    int id2 = id - 147456;
    int co = id2 / 1728, rem = id2 % 1728, r = rem >> 6, ci = rem & 63;
    wvp[id2] = (h16)wv[((size_t)co * 64 + ci) * 27 + r];
  }
}

// --- Kernel 2: implicit-GEMM conv, M=128 x N=128-pixel tile, K=KD*576 ------
// grid (ntile=8, l=Lout, b=8); block 256 = 8 waves in 4(M)x2(N), wave=32x64.
// B tile (32 cin x 128 pixels) built via on-the-fly im2col into LDS [n][ci].
// Software-pipelined: iteration it+1's global im2col loads are issued between
// the staging barrier and the WMMA block of iteration it.
// vmode=1 stores output transposed [b][l][hw][cout] (f16) for attn*V GEMM.
template <int KD>
__global__ __launch_bounds__(256) void conv_gemm(
    const float* __restrict__ x, const h16* __restrict__ wp,
    const float* __restrict__ bias, h16* __restrict__ out, float scale,
    int Lout, int vmode) {
  __shared__ h16 sB[128 * 40];  // [pixel][ci], stride 40 halves (16B aligned rows)
  const int tid = threadIdx.x, lane = tid & 31, wv = tid >> 5;
  const int mw = (wv & 3) * 32, nw = (wv >> 2) * 64;
  const int nt = blockIdx.x, l = blockIdx.y, b = blockIdx.z;
  const int n0 = nt * 128, h0 = nt * 4;
  const int K = KD * 576;
  const int kiter = KD * 18;

  const int ci_loc = tid >> 3, part = tid & 7;
  const int hh = part >> 1, w0 = (part & 1) * 16;
  const int nlb = hh * 32 + w0;

  v8f acc[2][4];
#pragma unroll
  for (int i = 0; i < 2; ++i)
#pragma unroll
    for (int j = 0; j < 4; ++j) zero8(acc[i][j]);

  const int ksA = (lane >> 4) * 16;
  const int ksB = (lane >> 4) * 16;

  // im2col register fetch for one K-chunk: 16 contiguous pixels of one row
  auto fetch = [&](int it, float (&v)[16]) {
    const int r = it >> 1, halfk = it & 1;
    const int dl = r / 9, rr = r % 9, kh = rr / 3, kw = rr % 3;
    const int ci = halfk * 32 + ci_loc;
    const int hs = h0 + hh + kh - 1;
    const bool okh = (unsigned)hs < 32u;
    const float* src =
        x + ((((size_t)b * CIN + ci) * LIN + (l + dl)) * 32 + (okh ? hs : 0)) * 32;
#pragma unroll
    for (int u = 0; u < 16; ++u) {
      const int wsrc = w0 + u + kw - 1;
      v[u] = (okh && (unsigned)wsrc < 32u) ? src[wsrc] : 0.0f;
    }
  };

  float stage[16];
  fetch(0, stage);

  for (int it = 0; it < kiter; ++it) {
#pragma unroll
    for (int u = 0; u < 16; ++u) sB[(nlb + u) * 40 + ci_loc] = (h16)stage[u];
    __syncthreads();

    float nxt[16];
    fetch((it + 1 < kiter) ? it + 1 : it, nxt);  // overlaps HBM with WMMA below

    v16h afr[2], bfr[4];
#pragma unroll
    for (int i = 0; i < 2; ++i) {  // A straight from packed weights in global
      const int m = mw + i * 16 + (lane & 15);
      afr[i] = frag_a16(wp + (size_t)m * K + it * 32 + ksA);
    }
#pragma unroll
    for (int j = 0; j < 4; ++j) {
      const int nn = nw + j * 16 + (lane & 15);
      bfr[j] = frag_b16(&sB[nn * 40 + ksB]);
    }
#pragma unroll
    for (int i = 0; i < 2; ++i)
#pragma unroll
      for (int j = 0; j < 4; ++j) acc[i][j] = wmma16(afr[i], bfr[j], acc[i][j]);
    __syncthreads();
#pragma unroll
    for (int u = 0; u < 16; ++u) stage[u] = nxt[u];
  }

  // epilogue: + bias, * scale (q uses 0.5), convert to f16, store
  const int mrow0 = (lane >> 4) * 8;
#pragma unroll
  for (int i = 0; i < 2; ++i) {
    const int mbase = mw + i * 16 + mrow0;
    v4f b0 = *(const v4f*)(bias + mbase);
    v4f b1 = *(const v4f*)(bias + mbase + 4);
    float bb[8];
#pragma unroll
    for (int e = 0; e < 8; ++e) bb[e] = (e < 4) ? b0[e] : b1[e - 4];
#pragma unroll
    for (int j = 0; j < 4; ++j) {
      const int nn = n0 + nw + j * 16 + (lane & 15);
      if (vmode) {  // v: [b][l][hw][cout], contiguous in cout -> one b128 store
        v8h tv;
#pragma unroll
        for (int e = 0; e < 8; ++e)
          tv[e] = (h16)((acc[i][j][e] + bb[e]) * scale);
        h16* dst = out + ((((size_t)b * 16 + l) * HWDIM + nn) * COUT + mbase);
        *(v8h*)dst = tv;
      } else {  // q/k: [b][c][l][hw] so K-dim of QK^T is contiguous
#pragma unroll
        for (int e = 0; e < 8; ++e) {
          const int m = mbase + e;
          out[(((size_t)b * COUT + m) * Lout + l) * HWDIM + nn] =
              (h16)((acc[i][j][e] + bb[e]) * scale);
        }
      }
    }
  }
}

// --- Kernel 3: split-K logits = Q·K^T per (b,head): 128x128, K=9*1024 ------
// grid (8 K-slices, 16 b*head) = 128 WGs. Each slice covers K=1152 (36 chunks)
// and writes a private partial buffer; softmax sums slices deterministically.
// All fragments read directly from global (K contiguous per row), no LDS.
__global__ __launch_bounds__(256) void attn_logits(const h16* __restrict__ q,
                                                   const h16* __restrict__ k,
                                                   float* __restrict__ lgp) {
  const int tid = threadIdx.x, lane = tid & 31, wv = tid >> 5;
  const int mw = (wv & 3) * 32, nw = (wv >> 2) * 64;
  const int ks = blockIdx.x;  // K-slice
  const int b2 = blockIdx.y, b = b2 >> 1, hd = b2 & 1;
  const size_t koff = (size_t)hd * 9216 + (size_t)ks * 1152;
  const h16* qb = q + (size_t)b * COUT * 18432 + koff;
  const h16* kb = k + (size_t)b * COUT * 18432 + koff;
  const int ksA = (lane >> 4) * 16, ksB = (lane >> 4) * 16;

  v8f acc[2][4];
#pragma unroll
  for (int i = 0; i < 2; ++i)
#pragma unroll
    for (int j = 0; j < 4; ++j) zero8(acc[i][j]);

  for (int it = 0; it < 36; ++it) {
    const int k0 = it * 32;
    v16h afr[2], bfr[4];
#pragma unroll
    for (int i = 0; i < 2; ++i) {
      const int row = mw + i * 16 + (lane & 15);
      afr[i] = frag_a16(qb + (size_t)row * 18432 + k0 + ksA);
    }
#pragma unroll
    for (int j = 0; j < 4; ++j) {
      const int m = nw + j * 16 + (lane & 15);
      bfr[j] = frag_b16(kb + (size_t)m * 18432 + k0 + ksB);
    }
#pragma unroll
    for (int i = 0; i < 2; ++i)
#pragma unroll
      for (int j = 0; j < 4; ++j) acc[i][j] = wmma16(afr[i], bfr[j], acc[i][j]);
  }

  float* lb = lgp + ((size_t)b2 * 8 + ks) * 16384;  // [b2][ks][c][m]
#pragma unroll
  for (int i = 0; i < 2; ++i)
#pragma unroll
    for (int j = 0; j < 4; ++j)
#pragma unroll
      for (int e = 0; e < 8; ++e) {
        const int c = mw + i * 16 + (lane >> 4) * 8 + e;
        const int m = nw + j * 16 + (lane & 15);
        lb[(size_t)c * 128 + m] = acc[i][j][e];
      }
}

// --- Kernel 4: fused split-K reduce + row softmax (2048 rows x 128) --------
// One wave per row; sums the 8 K-slice partials in fixed order (deterministic).
__global__ __launch_bounds__(256) void softmax_rows(
    const float* __restrict__ lgp, h16* __restrict__ attn) {
  const int lane = threadIdx.x & 31, wv = threadIdx.x >> 5;
  const int row = blockIdx.x * 8 + wv;  // row = b2*128 + c
  const int b2 = row >> 7, c = row & 127;
  v4f xv;
#pragma unroll
  for (int e = 0; e < 4; ++e) xv[e] = 0.0f;
#pragma unroll
  for (int ks = 0; ks < 8; ++ks)
    xv += *(const v4f*)(lgp + ((size_t)b2 * 8 + ks) * 16384 +
                        (size_t)c * 128 + lane * 4);
  float mx = fmaxf(fmaxf(xv[0], xv[1]), fmaxf(xv[2], xv[3]));
#pragma unroll
  for (int o = 16; o > 0; o >>= 1) mx = fmaxf(mx, __shfl_xor(mx, o, 32));
  float ex[4], s = 0.0f;
#pragma unroll
  for (int e = 0; e < 4; ++e) { ex[e] = __expf(xv[e] - mx); s += ex[e]; }
#pragma unroll
  for (int o = 16; o > 0; o >>= 1) s += __shfl_xor(s, o, 32);
  const float inv = 1.0f / s;
  v4h o4;
#pragma unroll
  for (int e = 0; e < 4; ++e) o4[e] = (h16)(ex[e] * inv);
  *(v4h*)(attn + (size_t)row * 128 + lane * 4) = o4;
}

// --- Kernel 5: out = attn · V per (b,head): 128 x 8192, K=128 --------------
// v stored [b][lv][hw][cout] -> B fragments contiguous in K, direct global.
__global__ __launch_bounds__(256) void attn_apply(const h16* __restrict__ attn,
                                                  const h16* __restrict__ vb,
                                                  float* __restrict__ outp) {
  const int tid = threadIdx.x, lane = tid & 31, wv = tid >> 5;
  const int mw = (wv & 3) * 32, nw = (wv >> 2) * 64;
  const int jt = blockIdx.x, b2 = blockIdx.y, b = b2 >> 1, hd = b2 & 1;
  const int d = jt >> 3, hw0 = (jt & 7) * 128, lv = hd * 8 + d;
  const h16* ab = attn + (size_t)b2 * 128 * 128;
  const h16* vbase = vb + (((size_t)b * 16 + lv) * HWDIM + hw0) * COUT;
  const int ksA = (lane >> 4) * 16, ksB = (lane >> 4) * 16;

  v8f acc[2][4];
#pragma unroll
  for (int i = 0; i < 2; ++i)
#pragma unroll
    for (int j = 0; j < 4; ++j) zero8(acc[i][j]);

#pragma unroll
  for (int it = 0; it < 4; ++it) {
    const int k0 = it * 32;
    v16h afr[2], bfr[4];
#pragma unroll
    for (int i = 0; i < 2; ++i) {
      const int row = mw + i * 16 + (lane & 15);
      afr[i] = frag_a16(ab + (size_t)row * 128 + k0 + ksA);
    }
#pragma unroll
    for (int j = 0; j < 4; ++j) {
      const int jloc = nw + j * 16 + (lane & 15);
      bfr[j] = frag_b16(vbase + (size_t)jloc * COUT + k0 + ksB);
    }
#pragma unroll
    for (int i = 0; i < 2; ++i)
#pragma unroll
      for (int j = 0; j < 4; ++j) acc[i][j] = wmma16(afr[i], bfr[j], acc[i][j]);
  }

  // out[b][c][hd*8+d][h][w], f32
#pragma unroll
  for (int i = 0; i < 2; ++i)
#pragma unroll
    for (int j = 0; j < 4; ++j)
#pragma unroll
      for (int e = 0; e < 8; ++e) {
        const int c = mw + i * 16 + (lane >> 4) * 8 + e;
        const int jloc = nw + j * 16 + (lane & 15);
        outp[(((size_t)b * COUT + c) * 16 + lv) * HWDIM + hw0 + jloc] =
            acc[i][j][e];
      }
}

// ---------------------------------------------------------------------------
extern "C" void kernel_launch(void* const* d_in, const int* in_sizes, int n_in,
                              void* d_out, int out_size, void* d_ws,
                              size_t ws_size, hipStream_t stream) {
  const float* input  = (const float*)d_in[0];
  const float* memory = (const float*)d_in[1];
  const float* wq = (const float*)d_in[2];
  const float* bq = (const float*)d_in[3];
  const float* wk = (const float*)d_in[4];
  const float* bk = (const float*)d_in[5];
  const float* wv = (const float*)d_in[6];
  const float* bv = (const float*)d_in[7];
  float* outp = (float*)d_out;

  // workspace layout (bytes, all 256B aligned)
  const size_t off_wqp = 0;                       // 128*576 f16
  const size_t off_wkp = off_wqp + 147456;        // 128*576 f16
  const size_t off_wvp = off_wkp + 147456;        // 128*1728 f16
  const size_t off_q   = off_wvp + 442368;        // 8*128*18*1024 f16
  const size_t off_k   = off_q + 37748736;
  const size_t off_v   = off_k + 37748736;        // 8*16*1024*128 f16 (transposed)
  const size_t off_lg  = off_v + 33554432;        // 16*8*128*128 f32 split-K partials
  const size_t off_at  = off_lg + 8388608;        // 16*128*128 f16
  const size_t total   = off_at + 524288;
  if (ws_size < total) return;  // harness scratch too small; nothing safe to do

  char* ws = (char*)d_ws;
  h16* wqp = (h16*)(ws + off_wqp);
  h16* wkp = (h16*)(ws + off_wkp);
  h16* wvp = (h16*)(ws + off_wvp);
  h16* qb  = (h16*)(ws + off_q);
  h16* kb  = (h16*)(ws + off_k);
  h16* vb  = (h16*)(ws + off_v);
  float* lg = (float*)(ws + off_lg);
  h16* at  = (h16*)(ws + off_at);

  pack_weights<<<1440, 256, 0, stream>>>(wq, wk, wv, wqp, wkp, wvp);
  conv_gemm<1><<<dim3(8, 18, 8), 256, 0, stream>>>(input,  wqp, bq, qb, 0.5f, 18, 0);
  conv_gemm<1><<<dim3(8, 18, 8), 256, 0, stream>>>(memory, wkp, bk, kb, 1.0f, 18, 0);
  conv_gemm<3><<<dim3(8, 16, 8), 256, 0, stream>>>(memory, wvp, bv, vb, 1.0f, 16, 1);
  attn_logits<<<dim3(8, 16), 256, 0, stream>>>(qb, kb, lg);
  softmax_rows<<<256, 256, 0, stream>>>(lg, at);
  attn_apply<<<dim3(64, 16), 256, 0, stream>>>(at, vb, outp);
}
